// CrossModalFusionModel_78726750535771
// MI455X (gfx1250) — matrix operand order repaired and checked by reading, not compiled, not run on hardware
//
#include <hip/hip_runtime.h>
#include <hip/hip_bf16.h>

// ---------------------------------------------------------------------------
// CDNA5 (gfx1250) cross-modal fusion:
//   txt = cap @ tp_w.T + tp_b ; img = images @ ip_w.T + ip_b
//   fused_img = SWA(img, txt, ia_*) ; fused_txt = SWA(txt, img, ta_*)
// GEMMs: v_wmma_f32_16x16x32_f16. Activations live as f16 -> A-tiles are
// staged into LDS with the async DMA engine (global_load_async_to_lds_b128,
// ASYNCcnt); weights are f32, converted to f16 in the LDS staging path and
// read exactly once. Softmax is f32 VALU, one wave per (query, head).
// ---------------------------------------------------------------------------

typedef __attribute__((ext_vector_type(4)))  _Float16 h4_t;
typedef __attribute__((ext_vector_type(8)))  _Float16 h8_t;
typedef __attribute__((ext_vector_type(16))) _Float16 h16_t;
typedef __attribute__((ext_vector_type(8)))  float    f8_t;

#define D_MODEL 1024
#define N_SEQ   512
#define HEADS   8
#define DH      128
#define WIN     64
#define LWIN    66

// ---------------------------------------------------------------------------
// f32 -> f16 convert (captions / images), n divisible by 1024
// ---------------------------------------------------------------------------
__global__ __launch_bounds__(256) void cvt_f32_to_f16_kernel(
    const float* __restrict__ in, _Float16* __restrict__ out, int n)
{
  const int i = (blockIdx.x * 256 + threadIdx.x) * 4;
  if (i < n) {
    const float4 v = *(const float4*)(in + i);
    h4_t h; h[0] = (_Float16)v.x; h[1] = (_Float16)v.y;
            h[2] = (_Float16)v.z; h[3] = (_Float16)v.w;
    *(h4_t*)(out + i) = h;
  }
}

// ---------------------------------------------------------------------------
// GEMM: out[n][m] = sum_k A[n][k] * W[m][k] + bias[m] (+ res[n][m])
// A: (512 x K) f16, W: (M x K) f32 row-major (torch Linear weight layout).
// Block tile 128x128, 8 waves (2x4), wave tile 64x32 -> 4x2 WMMA accumulators.
// BK = 32. A-tile staged via async DMA to LDS; W-tile converted f32->f16.
// ---------------------------------------------------------------------------
__global__ __launch_bounds__(256) void gemm_bias_res_kernel(
    const _Float16* __restrict__ A, const float* __restrict__ W,
    const float* __restrict__ bias, const float* __restrict__ res,
    float* __restrict__ out32, _Float16* __restrict__ out16, int K, int M)
{
  constexpr int LDA = 40;               // halves per LDS row (32 + 8 pad); 80B rows
  __shared__ _Float16 As[128 * LDA];
  __shared__ _Float16 Ws[128 * LDA];

  const int tid  = threadIdx.x;
  const int wave = tid >> 5;
  const int lane = tid & 31;
  const int wr   = wave >> 2;           // 0..1  (row group of 64)
  const int wc   = wave & 3;            // 0..3  (col group of 32)
  const int blockRow = blockIdx.x * 128;
  const int blockCol = blockIdx.y * 128;

  f8_t acc[4][2];
  #pragma unroll
  for (int r = 0; r < 4; ++r)
    #pragma unroll
    for (int c = 0; c < 2; ++c)
      #pragma unroll
      for (int j = 0; j < 8; ++j) acc[r][c][j] = 0.0f;

  const int m16 = lane & 15;
  const int hiHalf = (lane < 16) ? 0 : 1;
  const int kbA = hiHalf ? 8 : 0;       // A fragment K base (per ISA layout)
  const int kbB = hiHalf ? 8 : 0;       // B fragment K base

  for (int k0 = 0; k0 < K; k0 += 32) {
    // ---- A tile: raw f16 copy via async DMA engine (ASYNCcnt) -------------
    // 128 rows x 32 halves = 512 x 16B chunks; per-lane LDS dest keeps the
    // padded 80B row stride.
    #pragma unroll
    for (int j = 0; j < 2; ++j) {
      const int c   = tid + j * 256;    // 0..511
      const int row = c >> 2;
      const int q   = c & 3;            // 16B chunk within the 64B row
      const unsigned long long ga =
          (unsigned long long)(A + (size_t)(blockRow + row) * K + k0 + q * 8);
      const unsigned la = (unsigned)(unsigned long long)(&As[row * LDA + q * 8]);
      asm volatile("global_load_async_to_lds_b128 %0, %1, off"
                   :: "v"(la), "v"(ga) : "memory");
    }

    // ---- W tile: f32 global -> f16 LDS (VALU convert path) ----------------
    #pragma unroll
    for (int j = 0; j < 4; ++j) {
      const int idx = tid + j * 256;    // 1024 float4 slots cover 128x32
      const int row = idx >> 3;
      const int cv  = idx & 7;
      const float4 wv = *(const float4*)(W + (size_t)(blockCol + row) * K + k0 + cv * 4);
      h4_t wh; wh[0] = (_Float16)wv.x; wh[1] = (_Float16)wv.y;
               wh[2] = (_Float16)wv.z; wh[3] = (_Float16)wv.w;
      *(h4_t*)(&Ws[row * LDA + cv * 4]) = wh;
      if (j == 0 && k0 + 32 < K)        // pull next W K-tile toward L2/L0
        __builtin_prefetch(W + (size_t)(blockCol + row) * K + k0 + 32 + cv * 4, 0, 3);
    }

    asm volatile("s_wait_asynccnt 0x0" ::: "memory");
    __syncthreads();

    // ---- load fragments (documented CDNA5 wave32 layouts) -----------------
    h16_t afrag[4];
    #pragma unroll
    for (int r = 0; r < 4; ++r) {
      const int base = (wr * 64 + r * 16 + m16) * LDA + kbA;
      h8_t lo = *(const h8_t*)(&As[base]);        // K = kbA .. kbA+7
      h8_t hi = *(const h8_t*)(&As[base + 16]);   // K = kbA+16 .. kbA+23
      afrag[r] = __builtin_shufflevector(lo, hi, 0,1,2,3,4,5,6,7,8,9,10,11,12,13,14,15);
    }
    h16_t bfrag[2];
    #pragma unroll
    for (int c = 0; c < 2; ++c) {
      const int base = (wc * 32 + c * 16 + m16) * LDA + kbB;
      h8_t lo = *(const h8_t*)(&Ws[base]);        // K = kbB .. kbB+7
      h8_t hi = *(const h8_t*)(&Ws[base + 16]);   // K = kbB+16 .. kbB+23
      bfrag[c] = __builtin_shufflevector(lo, hi, 0,1,2,3,4,5,6,7,8,9,10,11,12,13,14,15);
    }

    // ---- 8 WMMA issues per wave per K-step --------------------------------
    #pragma unroll
    for (int r = 0; r < 4; ++r)
      #pragma unroll
      for (int c = 0; c < 2; ++c)
        acc[r][c] = __builtin_amdgcn_wmma_f32_16x16x32_f16(
            false, afrag[r], false, bfrag[c], (short)0, acc[r][c], false, false);

    __syncthreads();
  }

  // ---- epilogue: bias (+ residual), C layout: VGPR j -> M = j + 8*hiHalf --
  #pragma unroll
  for (int c = 0; c < 2; ++c) {
    const int n = blockCol + wc * 32 + c * 16 + m16;
    const float b = bias[n];
    #pragma unroll
    for (int r = 0; r < 4; ++r) {
      const int mBase = blockRow + wr * 64 + r * 16 + (hiHalf ? 8 : 0);
      #pragma unroll
      for (int j = 0; j < 8; ++j) {
        const size_t o = (size_t)(mBase + j) * M + n;
        float v = acc[r][c][j] + b;
        if (res) v += res[o];
        if (out32) out32[o] = v;
        if (out16) out16[o] = (_Float16)v;
      }
    }
  }
}

// ---------------------------------------------------------------------------
// Sliding-window attention core (post Q/K/V projection).
// One wave per (query i, head h). L = 66 slots; valid slots read K/V rows,
// zero-pad slots (reference quirk) attend against the raw biases kb/vb,
// fully-masked slots are skipped (== -1e9 score -> exp 0).
// Output written as f16 (feeds the async A path of the output projection).
// ---------------------------------------------------------------------------
__global__ __launch_bounds__(256) void swa_attn_kernel(
    const float* __restrict__ Q, const float* __restrict__ Kf,
    const float* __restrict__ Vf, const float* __restrict__ kb,
    const float* __restrict__ vb, _Float16* __restrict__ O)
{
  const int i    = blockIdx.x;          // query token
  const int h    = threadIdx.x >> 5;    // head = wave id
  const int lane = threadIdx.x & 31;
  const int d0   = lane * 4;            // 4 of dh=128 per lane

  const float4 q4 = *(const float4*)(Q + (size_t)i * D_MODEL + h * DH + d0);

  // zero-pad bookkeeping, faithful to the reference mask construction
  const int lo = (32 - i) > 0 ? (32 - i) : 0;          // leading invalid slots
  const int hi = (i - 478) > 0 ? (i - 478) : 0;        // trailing invalid slots
  const int length = LWIN - lo - hi;
  const int nz = (WIN - length) > 0 ? (WIN - length) : 0;

  float m = -INFINITY, s = 0.0f;
  float ax = 0.0f, ay = 0.0f, az = 0.0f, aw = 0.0f;

  for (int l = 0; l < LWIN; ++l) {
    const int idx = i + l - 32;
    const float* kp;
    const float* vp;
    if (idx >= 0 && idx < N_SEQ) {
      kp = Kf + (size_t)idx * D_MODEL + h * DH;
      vp = Vf + (size_t)idx * D_MODEL + h * DH;
    } else {
      const int rank = (idx < 0) ? l : (lo + l - (LWIN - hi));
      if (rank >= nz) continue;         // fully masked slot (uniform per wave)
      kp = kb + h * DH;                 // zero context row -> k = kb, v = vb
      vp = vb + h * DH;
    }
    const float4 k4 = *(const float4*)(kp + d0);
    float p = q4.x * k4.x + q4.y * k4.y + q4.z * k4.z + q4.w * k4.w;
    #pragma unroll
    for (int off = 16; off >= 1; off >>= 1) p += __shfl_xor(p, off, 32);
    p *= 0.08838834764831845f;          // 1/sqrt(128)

    const float nm = fmaxf(m, p);
    const float sc = __expf(m - nm);    // exp(-inf)=0 on first attended slot
    const float e  = __expf(p - nm);
    s = s * sc + e;
    const float4 v4 = *(const float4*)(vp + d0);
    ax = ax * sc + e * v4.x;
    ay = ay * sc + e * v4.y;
    az = az * sc + e * v4.z;
    aw = aw * sc + e * v4.w;
    m = nm;
  }

  const float inv = 1.0f / s;
  h4_t o4; o4[0] = (_Float16)(ax * inv); o4[1] = (_Float16)(ay * inv);
           o4[2] = (_Float16)(az * inv); o4[3] = (_Float16)(aw * inv);
  *(h4_t*)(O + (size_t)i * D_MODEL + h * DH + d0) = o4;
}

// ---------------------------------------------------------------------------
extern "C" void kernel_launch(void* const* d_in, const int* in_sizes, int n_in,
                              void* d_out, int out_size, void* d_ws, size_t ws_size,
                              hipStream_t stream) {
  (void)in_sizes; (void)n_in; (void)out_size; (void)ws_size;

  const float* images   = (const float*)d_in[0];   // 512 x 1024
  const float* captions = (const float*)d_in[1];   // 512 x 768
  const float* tp_w = (const float*)d_in[3];  const float* tp_b = (const float*)d_in[4];
  const float* ip_w = (const float*)d_in[5];  const float* ip_b = (const float*)d_in[6];
  const float* ia_qw = (const float*)d_in[7];  const float* ia_qb = (const float*)d_in[8];
  const float* ia_kw = (const float*)d_in[9];  const float* ia_kb = (const float*)d_in[10];
  const float* ia_vw = (const float*)d_in[11]; const float* ia_vb = (const float*)d_in[12];
  const float* ia_ow = (const float*)d_in[13]; const float* ia_ob = (const float*)d_in[14];
  const float* ta_qw = (const float*)d_in[15]; const float* ta_qb = (const float*)d_in[16];
  const float* ta_kw = (const float*)d_in[17]; const float* ta_kb = (const float*)d_in[18];
  const float* ta_vw = (const float*)d_in[19]; const float* ta_vb = (const float*)d_in[20];
  const float* ta_ow = (const float*)d_in[21]; const float* ta_ob = (const float*)d_in[22];

  float* fused_img = (float*)d_out;
  float* fused_txt = (float*)d_out + (size_t)N_SEQ * D_MODEL;

  // ~15 MB workspace, L2-resident (192 MB L2)
  char* ws = (char*)d_ws;
  size_t off = 0;
  auto carve = [&](size_t bytes) { void* p = ws + off; off += (bytes + 255) & ~(size_t)255; return p; };
  _Float16* capH    = (_Float16*)carve((size_t)N_SEQ * 768 * 2);
  _Float16* imgRawH = (_Float16*)carve((size_t)N_SEQ * D_MODEL * 2);
  float*    txtF    = (float*)   carve((size_t)N_SEQ * D_MODEL * 4);
  _Float16* txtH    = (_Float16*)carve((size_t)N_SEQ * D_MODEL * 2);
  float*    imgF    = (float*)   carve((size_t)N_SEQ * D_MODEL * 4);
  _Float16* imgH    = (_Float16*)carve((size_t)N_SEQ * D_MODEL * 2);
  float*    Qf      = (float*)   carve((size_t)N_SEQ * D_MODEL * 4);
  float*    Kf      = (float*)   carve((size_t)N_SEQ * D_MODEL * 4);
  float*    Vf      = (float*)   carve((size_t)N_SEQ * D_MODEL * 4);
  _Float16* OH      = (_Float16*)carve((size_t)N_SEQ * D_MODEL * 2);

  const dim3 gGrid(N_SEQ / 128, D_MODEL / 128);    // 4 x 8
  const dim3 gBlk(256);
  const dim3 aGrid(N_SEQ);
  const dim3 aBlk(256);

  // f16 copies of the raw modalities
  cvt_f32_to_f16_kernel<<<dim3((N_SEQ * 768) / 1024), gBlk, 0, stream>>>(captions, capH, N_SEQ * 768);
  cvt_f32_to_f16_kernel<<<dim3((N_SEQ * D_MODEL) / 1024), gBlk, 0, stream>>>(images, imgRawH, N_SEQ * D_MODEL);

  // input projections (emit both f32 and f16 views)
  gemm_bias_res_kernel<<<gGrid, gBlk, 0, stream>>>(capH,    tp_w, tp_b, nullptr, txtF, txtH, 768,  D_MODEL);
  gemm_bias_res_kernel<<<gGrid, gBlk, 0, stream>>>(imgRawH, ip_w, ip_b, nullptr, imgF, imgH, 1024, D_MODEL);

  // attention 1: x = img, ctx = txt (K/V projected ONCE per ctx token)
  gemm_bias_res_kernel<<<gGrid, gBlk, 0, stream>>>(imgH, ia_qw, ia_qb, nullptr, Qf, nullptr, 1024, D_MODEL);
  gemm_bias_res_kernel<<<gGrid, gBlk, 0, stream>>>(txtH, ia_kw, ia_kb, nullptr, Kf, nullptr, 1024, D_MODEL);
  gemm_bias_res_kernel<<<gGrid, gBlk, 0, stream>>>(txtH, ia_vw, ia_vb, nullptr, Vf, nullptr, 1024, D_MODEL);
  swa_attn_kernel<<<aGrid, aBlk, 0, stream>>>(Qf, Kf, Vf, ia_kb, ia_vb, OH);
  gemm_bias_res_kernel<<<gGrid, gBlk, 0, stream>>>(OH, ia_ow, ia_ob, imgF, fused_img, nullptr, 1024, D_MODEL);

  // attention 2: x = txt, ctx = img
  gemm_bias_res_kernel<<<gGrid, gBlk, 0, stream>>>(txtH, ta_qw, ta_qb, nullptr, Qf, nullptr, 1024, D_MODEL);
  gemm_bias_res_kernel<<<gGrid, gBlk, 0, stream>>>(imgH, ta_kw, ta_kb, nullptr, Kf, nullptr, 1024, D_MODEL);
  gemm_bias_res_kernel<<<gGrid, gBlk, 0, stream>>>(imgH, ta_vw, ta_vb, nullptr, Vf, nullptr, 1024, D_MODEL);
  swa_attn_kernel<<<aGrid, aBlk, 0, stream>>>(Qf, Kf, Vf, ta_kb, ta_vb, OH);
  gemm_bias_res_kernel<<<gGrid, gBlk, 0, stream>>>(OH, ta_ow, ta_ob, txtF, fused_txt, nullptr, 1024, D_MODEL);
}
